// LinearAttentionVit_32521492365371
// MI455X (gfx1250) — matrix operand order repaired and checked by reading, not compiled, not run on hardware
//
#include <hip/hip_runtime.h>
#include <hip/hip_bf16.h>

typedef unsigned short u16;
typedef __attribute__((ext_vector_type(16))) __bf16 v16bf;
typedef __attribute__((ext_vector_type(8)))  float  v8f;

// ---------- constants for this problem ----------
#define BB 64
#define NN 1024
#define CC 768
#define HH 12
#define DD 64
#define M_ROWS (BB * NN)        // 65536
#define N_QKV  (3 * CC)         // 2304
#define OUT_ELEMS ((size_t)M_ROWS * CC)   // 50331648 (then attn, same count)

// ---------- fp32 <-> bf16 helpers ----------
__device__ __forceinline__ u16 f2bf(float f) {
    union { float f; unsigned u; } x; x.f = f;
    unsigned u = x.u;
    unsigned r = u + 0x7FFFu + ((u >> 16) & 1u);   // round-to-nearest-even
    return (u16)(r >> 16);
}
__device__ __forceinline__ float bf2f(u16 h) {
    union { unsigned u; float f; } x; x.u = ((unsigned)h) << 16;
    return x.f;
}

// ---------- stage 1: conversions ----------
__global__ void cvt_f32_bf16(const float* __restrict__ in, u16* __restrict__ out, int n) {
    int i = blockIdx.x * blockDim.x + threadIdx.x;
    int stride = gridDim.x * blockDim.x;
    for (; i < n; i += stride) out[i] = f2bf(in[i]);
}

// in: R x Cc row-major (K x N). out: Cc x R row-major (N x K), bf16.
__global__ void transpose_cvt(const float* __restrict__ in, u16* __restrict__ out, int R, int Cc) {
    int i = blockIdx.x * blockDim.x + threadIdx.x;
    int total = R * Cc;
    int stride = gridDim.x * blockDim.x;
    for (; i < total; i += stride) {
        int k = i % R;
        int n = i / R;
        out[i] = f2bf(in[(size_t)k * Cc + n]);   // out[n*R + k]
    }
}

// ---------- stage 2/5: bf16 WMMA GEMM  C[M,N] = A[M,K] * Bt[N,K]^T ----------
// A row-major bf16, Bt row-major bf16 (i.e. B stored transposed).
// Block = 256 threads = 8 waves; block tile 128(M) x 64(N); each wave: 16x64.
template <bool OUT_BF16, bool ADD_BIAS>
__global__ __launch_bounds__(256) void gemm_bf16_wmma(
    const u16* __restrict__ A, const u16* __restrict__ Bt,
    void* __restrict__ Cout, const float* __restrict__ bias,
    int M, int Nn, int K)
{
    const int lane = threadIdx.x & 31;
    const int wave = threadIdx.x >> 5;
    const int m0 = blockIdx.y * 128 + wave * 16;
    const int n0 = blockIdx.x * 64;

    const int mrow  = m0 + (lane & 15);
    const int khalfA = (lane >> 4) * 8;    // unused; A frag also uses khalfB: K in [khalfB, khalfB+16), contiguous
    const int khalfB = (lane >> 4) * 16;   // B frag: K in [khalfB, khalfB+16), contiguous

    v8f zero = {};
    v8f acc0 = zero, acc1 = zero, acc2 = zero, acc3 = zero;

    const u16* Arow = A + (size_t)mrow * K;
    const u16* Brow0 = Bt + (size_t)(n0 +  0 + (lane & 15)) * K;
    const u16* Brow1 = Bt + (size_t)(n0 + 16 + (lane & 15)) * K;
    const u16* Brow2 = Bt + (size_t)(n0 + 32 + (lane & 15)) * K;
    const u16* Brow3 = Bt + (size_t)(n0 + 48 + (lane & 15)) * K;

    for (int k0 = 0; k0 < K; k0 += 32) {
        __builtin_prefetch(Arow + k0 + 64, 0, 1);   // global_prefetch for next tiles

        union { v16bf v; uint4 q[2]; } af;
        af.q[0] = *(const uint4*)(Arow + k0 + khalfB);
        af.q[1] = *(const uint4*)(Arow + k0 + khalfB + 8);

        union { v16bf v; uint4 q[2]; } bf0, bf1, bf2, bf3;
        bf0.q[0] = *(const uint4*)(Brow0 + k0 + khalfB);
        bf0.q[1] = *(const uint4*)(Brow0 + k0 + khalfB + 8);
        bf1.q[0] = *(const uint4*)(Brow1 + k0 + khalfB);
        bf1.q[1] = *(const uint4*)(Brow1 + k0 + khalfB + 8);
        bf2.q[0] = *(const uint4*)(Brow2 + k0 + khalfB);
        bf2.q[1] = *(const uint4*)(Brow2 + k0 + khalfB + 8);
        bf3.q[0] = *(const uint4*)(Brow3 + k0 + khalfB);
        bf3.q[1] = *(const uint4*)(Brow3 + k0 + khalfB + 8);

        acc0 = __builtin_amdgcn_wmma_f32_16x16x32_bf16(false, af.v, false, bf0.v, (short)0, acc0, false, false);
        acc1 = __builtin_amdgcn_wmma_f32_16x16x32_bf16(false, af.v, false, bf1.v, (short)0, acc1, false, false);
        acc2 = __builtin_amdgcn_wmma_f32_16x16x32_bf16(false, af.v, false, bf2.v, (short)0, acc2, false, false);
        acc3 = __builtin_amdgcn_wmma_f32_16x16x32_bf16(false, af.v, false, bf3.v, (short)0, acc3, false, false);
    }

    // C layout: VGPR r -> row m0 + r + 8*(lane>>4); column n0 + j*16 + (lane&15)
    const int rowb = m0 + 8 * (lane >> 4);
    const int col0 = lane & 15;
    v8f accs[4] = {acc0, acc1, acc2, acc3};
#pragma unroll
    for (int j = 0; j < 4; ++j) {
        int col = n0 + j * 16 + col0;
        float bv = ADD_BIAS ? bias[col] : 0.0f;
#pragma unroll
        for (int r = 0; r < 8; ++r) {
            size_t idx = (size_t)(rowb + r) * Nn + col;
            float val = accs[j][r] + bv;
            if (OUT_BF16) ((u16*)Cout)[idx] = f2bf(val);
            else          ((float*)Cout)[idx] = val;
        }
    }
}

// ---------- stage 3: per-(b,h,d) column statistics over N ----------
// qkv: bf16, layout (B,N,3,H,D) flattened. One block per (b,h).
__global__ __launch_bounds__(256) void stats_kernel(
    const u16* __restrict__ qkv, float* __restrict__ cA, float* __restrict__ cY)
{
    int bh = blockIdx.x;
    int h = bh % HH, b = bh / HH;
    int t = threadIdx.x;
    int d = t & 63, part = t >> 6;   // 4 partial sums per column

    float sq = 0.f, sq2 = 0.f, sk = 0.f, sk2 = 0.f;
    for (int n = part; n < NN; n += 4) {
        size_t base = ((size_t)(b * NN + n) * 3) * CC + h * DD + d;
        float qf = bf2f(qkv[base]);
        float kf = bf2f(qkv[base + CC]);
        sq += qf; sq2 += qf * qf; sk += kf; sk2 += kf * kf;
    }
    __shared__ float s[256 * 4];
    s[t * 4 + 0] = sq; s[t * 4 + 1] = sq2; s[t * 4 + 2] = sk; s[t * 4 + 3] = sk2;
    __syncthreads();
    if (t < 64) {
        float Sq = 0.f, Sq2 = 0.f, Sk = 0.f, Sk2 = 0.f;
        for (int p = 0; p < 4; ++p) {
            int idx = (p * 64 + t) * 4;
            Sq += s[idx]; Sq2 += s[idx + 1]; Sk += s[idx + 2]; Sk2 += s[idx + 3];
        }
        float rq = rsqrtf(Sq2), rk = rsqrtf(Sk2);
        float dinv = 1.0f / ((float)NN + (Sq * rq) * (Sk * rk));
        cA[bh * DD + t] = rq * rk * dinv;       // attn = q*k*cA
        cY[bh * DD + t] = rq * dinv / Sk2;      // y    = v*k*k*q*cY
    }
}

// ---------- stage 4: fused elementwise: attn (f32, B,H,D,N) + y (bf16, B,N,C) ----------
// One block per (b,h, n-tile of 64). LDS transpose so both sides coalesce.
__global__ __launch_bounds__(256) void fuse_elemwise(
    const u16* __restrict__ qkv, const float* __restrict__ cA, const float* __restrict__ cY,
    float* __restrict__ attn, u16* __restrict__ yb)
{
    int blk = blockIdx.x;
    int nt = blk & 15;           // N/64 = 16 tiles
    int bh = blk >> 4;
    int h = bh % HH, b = bh / HH;
    int n0 = nt * 64;

    __shared__ float tile[64 * 65];
    int t = threadIdx.x;
    int d = t & 63, ng = t >> 6;
    float ca = cA[bh * DD + d];
    float cy = cY[bh * DD + d];

#pragma unroll 4
    for (int i = 0; i < 16; ++i) {
        int nl = ng * 16 + i;
        int n = n0 + nl;
        size_t base = ((size_t)(b * NN + n) * 3) * CC + h * DD + d;
        float qf = bf2f(qkv[base]);
        float kf = bf2f(qkv[base + CC]);
        float vf = bf2f(qkv[base + 2 * CC]);
        tile[d * 65 + nl] = qf * kf * ca;
        yb[(size_t)(b * NN + n) * CC + h * DD + d] = f2bf(vf * kf * kf * qf * cy);
    }
    __syncthreads();
    int nl2 = t & 63, dg = t >> 6;
#pragma unroll 4
    for (int i = 0; i < 16; ++i) {
        int dd = dg * 16 + i;
        attn[((size_t)bh * DD + dd) * NN + n0 + nl2] = tile[dd * 65 + nl2];
    }
}

// ---------- launcher ----------
extern "C" void kernel_launch(void* const* d_in, const int* in_sizes, int n_in,
                              void* d_out, int out_size, void* d_ws, size_t ws_size,
                              hipStream_t stream) {
    (void)in_sizes; (void)n_in; (void)out_size; (void)ws_size;
    const float* x      = (const float*)d_in[0];   // (B,N,C)
    const float* w_qkv  = (const float*)d_in[1];   // (C, 3C)
    const float* w_proj = (const float*)d_in[2];   // (C, C)
    const float* b_proj = (const float*)d_in[3];   // (C,)

    float* outp  = (float*)d_out;                  // (B,N,C)
    float* attnp = outp + OUT_ELEMS;               // (B,H,D,N)

    // workspace layout (all 256B aligned)
    char* ws = (char*)d_ws;
    u16*   xb     = (u16*)(ws);                                       // 100,663,296 B
    u16*   qkvb   = (u16*)(ws + 100663296ull);                        // 301,989,888 B
    u16*   yb     = (u16*)(ws + 402653184ull);                        // 100,663,296 B
    u16*   wqkvT  = (u16*)(ws + 503316480ull);                        //   3,538,944 B
    u16*   wprojT = (u16*)(ws + 506855424ull);                        //   1,179,648 B
    float* cA     = (float*)(ws + 508035072ull);                      //     196,608 B
    float* cY     = (float*)(ws + 508231680ull);                      //     196,608 B

    // 1) conversions
    cvt_f32_bf16<<<4096, 256, 0, stream>>>(x, xb, (int)(M_ROWS * (size_t)CC));
    transpose_cvt<<<2048, 256, 0, stream>>>(w_qkv, wqkvT, CC, N_QKV);   // -> (3C x C)
    transpose_cvt<<<1024, 256, 0, stream>>>(w_proj, wprojT, CC, CC);    // -> (C x C)

    // 2) qkv = x @ w_qkv   (bf16 out)
    {
        dim3 grid(N_QKV / 64, M_ROWS / 128);
        gemm_bf16_wmma<true, false><<<grid, 256, 0, stream>>>(
            xb, wqkvT, (void*)qkvb, nullptr, M_ROWS, N_QKV, CC);
    }

    // 3) column stats -> cA, cY
    stats_kernel<<<BB * HH, 256, 0, stream>>>(qkvb, cA, cY);

    // 4) attn (f32) + y (bf16)
    fuse_elemwise<<<BB * HH * (NN / 64), 256, 0, stream>>>(qkvb, cA, cY, attnp, yb);

    // 5) out = y @ w_proj + b_proj   (f32 out)
    {
        dim3 grid(CC / 64, M_ROWS / 128);
        gemm_bf16_wmma<false, true><<<grid, 256, 0, stream>>>(
            yb, wprojT, (void*)outp, b_proj, M_ROWS, CC, CC);
    }
}
